// Gemma2Model_58978490908724
// MI455X (gfx1250) — matrix operand order repaired and checked by reading, not compiled, not run on hardware
//
#include <hip/hip_runtime.h>
#include <math.h>

// ---------------------------------------------------------------------------
// Gemma2 (2-layer) forward for gfx1250 / MI455X.
// bf16 WMMA (v_wmma_f32_16x16x32_bf16) for all GEMMs + attention, fp32 accum.
// fp32 weights converted to bf16 while staging into LDS (weights are read
// exactly once -> HBM-bound; no separate conversion pass). K/A tiles staged
// with CDNA5 async global->LDS DMA (ASYNCcnt); V staged transposed so P*V
// fragments are contiguous ds_load_b128. Flash-style online-softmax attention.
// ---------------------------------------------------------------------------

static constexpr int   kB   = 2;
static constexpr int   kS   = 2048;
static constexpr int   kHID = 2048;
static constexpr int   kNH  = 8;
static constexpr int   kNKV = 4;
static constexpr int   kHD  = 256;
static constexpr int   kFF  = 8192;
static constexpr int   kL   = 2;
static constexpr int   kTok = kB * kS;           // 4096 tokens
static constexpr float kScale   = 0.0625f;       // 256^-0.5
static constexpr float kSoftcap = 50.0f;
static constexpr float kCapMul  = 0.0625f / 50.0f;   // SCALE / SOFTCAP
static constexpr float kEps     = 1e-6f;

typedef __attribute__((ext_vector_type(16))) __bf16        bf16x16;
typedef __attribute__((ext_vector_type(8)))  float         f32x8;
typedef __attribute__((ext_vector_type(4)))  float         f32x4;
typedef __attribute__((ext_vector_type(4)))  unsigned int  u32x4;

union Frag {
  bf16x16        v;
  u32x4          q[2];
  unsigned short s[16];
};

__device__ __forceinline__ unsigned short f2bf(float x) {
  __bf16 b = (__bf16)x;                       // native cvt, RNE
  return __builtin_bit_cast(unsigned short, b);
}
__device__ __forceinline__ float bf2f(unsigned short u) {
  return (float)__builtin_bit_cast(__bf16, u);
}
__device__ __forceinline__ f32x8 wmma_bf16(const Frag& a, const Frag& b, f32x8 c) {
  // D = A(16x32 bf16) * B(32x16 bf16) + C(16x16 f32)
  return __builtin_amdgcn_wmma_f32_16x16x32_bf16(
      false, a.v, false, b.v, (short)0, c, false, false);
}
// overflow-safe tanh from v_exp_f32
__device__ __forceinline__ float fast_tanh(float x) {
  const float a = fabsf(x);
  const float e = __expf(2.0f * a);
  const float t = 1.0f - 2.0f / (e + 1.0f);
  return copysignf(t, x);
}

// CDNA5 async global->LDS copy (ASYNCcnt-tracked), 16B per lane.
__device__ __forceinline__ void async_copy_b128(void* lds_ptr, const void* gptr) {
  const unsigned lds_off = (unsigned)(unsigned long long)lds_ptr;  // low 32 bits = LDS offset
  asm volatile("global_load_async_to_lds_b128 %0, %1, off"
               :: "v"(lds_off), "v"(gptr)
               : "memory");
}
__device__ __forceinline__ void wait_async() {
  asm volatile("s_wait_asynccnt 0" ::: "memory");
}

// ---------------------------------------------------------------------------
// Embedding gather: h[tok, :] = emb[id] * sqrt(HID)
// ---------------------------------------------------------------------------
__global__ __launch_bounds__(256) void embed_kernel(const int* __restrict__ ids,
                                                    const float* __restrict__ emb,
                                                    float* __restrict__ H) {
  const long long tok = blockIdx.x;
  const long long id  = ids[tok];
  const float*    e   = emb + id * (long long)kHID;
  float*          h   = H + tok * (long long)kHID;
  const float     sc  = 45.25483399593904f;   // sqrt(2048)
  for (int c = threadIdx.x; c < kHID; c += 256) h[c] = e[c] * sc;
}

// ---------------------------------------------------------------------------
// RMSNorm. MODE 0: bf16 out = rms(x,w); MODE 1: f32 out = rms(x,w);
//          MODE 2: f32 out = x + rms(x,w)   (Gemma2 post-attn combine)
// One block per row of 2048; 8 elements cached per thread.
// ---------------------------------------------------------------------------
template <int MODE>
__global__ __launch_bounds__(256) void rmsnorm_kernel(const float* __restrict__ X,
                                                      const float* __restrict__ Wn,
                                                      void* __restrict__ Y) {
  __shared__ float red[8];
  const long long row = blockIdx.x;
  const float*    x   = X + row * (long long)kHID;
  float v[8];
  float ss = 0.f;
#pragma unroll
  for (int i = 0; i < 8; ++i) {
    v[i] = x[threadIdx.x + 256 * i];
    ss += v[i] * v[i];
  }
#pragma unroll
  for (int m = 1; m < 32; m <<= 1) ss += __shfl_xor(ss, m, 32);
  if ((threadIdx.x & 31) == 0) red[threadIdx.x >> 5] = ss;
  __syncthreads();
  float tot = 0.f;
#pragma unroll
  for (int i = 0; i < 8; ++i) tot += red[i];
  const float inv = rsqrtf(tot * (1.0f / (float)kHID) + kEps);
#pragma unroll
  for (int i = 0; i < 8; ++i) {
    const int c = threadIdx.x + 256 * i;
    const float yv = v[i] * inv * (1.0f + Wn[c]);
    const long long idx = row * (long long)kHID + c;
    if (MODE == 0)      ((unsigned short*)Y)[idx] = f2bf(yv);
    else if (MODE == 1) ((float*)Y)[idx] = yv;
    else                ((float*)Y)[idx] = v[i] + yv;
  }
}

// ---------------------------------------------------------------------------
// RoPE in-place on bf16 q [tok, NH*HD] and k [tok, NKV*HD].
// ---------------------------------------------------------------------------
__global__ __launch_bounds__(256) void rope_kernel(unsigned short* __restrict__ Qm,
                                                   unsigned short* __restrict__ Km,
                                                   const int* __restrict__ pos_ids) {
  const long long tok = blockIdx.x;
  const float     pos = (float)pos_ids[tok];
  for (int it = threadIdx.x; it < (kNH + kNKV) * (kHD / 2); it += 256) {
    unsigned short* base;
    int j;
    if (it < kNH * (kHD / 2)) {
      const int head = it >> 7; j = it & 127;
      base = Qm + tok * (long long)(kNH * kHD) + head * kHD;
    } else {
      const int r = it - kNH * (kHD / 2);
      const int head = r >> 7; j = r & 127;
      base = Km + tok * (long long)(kNKV * kHD) + head * kHD;
    }
    // 10000^(-j/128) = exp2(-j * log2(10000)/128)
    const float invf = exp2f(-(float)j * (13.287712379549449f / 128.0f));
    const float fr = pos * invf;
    const float c = cosf(fr), s = sinf(fr);
    const float x1 = bf2f(base[j]);
    const float x2 = bf2f(base[j + 128]);
    base[j]       = f2bf(x1 * c - x2 * s);
    base[j + 128] = f2bf(x2 * c + x1 * s);
  }
}

// ---------------------------------------------------------------------------
// Generic bf16 WMMA GEMM: C[M,N] = A[M,K](bf16,row-major) * Bw[K,N](f32).
// Block = 256 threads (8 waves), tile BM=BN=128, BK=32.
// Wave (wrow,wcol) owns a 32x64 tile = 2x4 WMMA accumulators.
// A tile staged with async global->LDS DMA; B tile converted f32->bf16 and
// stored transposed (two k-rows packed per ds_store_b32) so fragments are
// contiguous 16B ds reads per the ISA layouts.
// ---------------------------------------------------------------------------
enum { EPI_BF16 = 0, EPI_F32 = 1, EPI_GELU_MUL = 2, EPI_ADD_F32 = 3 };

template <int EPI>
__global__ __launch_bounds__(256) void gemm_kernel(const unsigned short* __restrict__ A,
                                                   const float* __restrict__ Bw,
                                                   void* __restrict__ Cout,
                                                   const void* __restrict__ aux,
                                                   int M, int N, int K) {
  __shared__ unsigned short As[128][32];    // A tile, row-major
  __shared__ unsigned short BsT[128][32];   // B tile, transposed: BsT[n][k]

  const int t    = threadIdx.x;
  const int lane = t & 31, w = t >> 5;
  const int wrow = w >> 1, wcol = w & 1;
  const int half = lane >> 4, lr = lane & 15;
  const long long bM = (long long)blockIdx.y * 128;
  const long long bN = (long long)blockIdx.x * 128;

  f32x8 acc[2][4];
#pragma unroll
  for (int mi = 0; mi < 2; ++mi)
#pragma unroll
    for (int ni = 0; ni < 4; ++ni)
#pragma unroll
      for (int r = 0; r < 8; ++r) acc[mi][ni][r] = 0.f;

  const int arow = t >> 1, acol = (t & 1) * 16;    // A: 128 rows x 32 cols bf16
  const int bkp  = t >> 4;                         // B: k-row pair 0..15
  const int bcol = (t & 15) * 8;                   //    8 cols per thread

  for (int k0 = 0; k0 < K; k0 += 32) {
    // stage A via async DMA (2 x 16B per thread)
    const unsigned short* ap = A + (bM + arow) * (long long)K + k0 + acol;
    async_copy_b128(&As[arow][acol], ap);
    async_copy_b128(&As[arow][acol + 8], ap + 8);
    // stage B: two k-rows, convert f32->bf16, pack pairs along k
    {
      const float* b0 = Bw + (long long)(k0 + 2 * bkp) * N + bN + bcol;
      const float* b1 = b0 + N;
      const f32x4 r0a = *(const f32x4*)b0,       r0b = *(const f32x4*)(b0 + 4);
      const f32x4 r1a = *(const f32x4*)b1,       r1b = *(const f32x4*)(b1 + 4);
#pragma unroll
      for (int i = 0; i < 4; ++i) {
        const unsigned p0 = (unsigned)f2bf(r0a[i]) | ((unsigned)f2bf(r1a[i]) << 16);
        const unsigned p1 = (unsigned)f2bf(r0b[i]) | ((unsigned)f2bf(r1b[i]) << 16);
        *(unsigned*)&BsT[bcol + i][2 * bkp]     = p0;
        *(unsigned*)&BsT[bcol + 4 + i][2 * bkp] = p1;
      }
    }
    wait_async();
    __syncthreads();

    Frag afr[2], bfr[4];
#pragma unroll
    for (int mi = 0; mi < 2; ++mi) {
      const int mr = wrow * 32 + mi * 16 + lr;
      // 16-bit A 16x32 layout: half 0 -> K{0..7,16..23}, half 1 -> K{8..15,24..31}
      afr[mi].q[0] = *(const u32x4*)&As[mr][half * 8];
      afr[mi].q[1] = *(const u32x4*)&As[mr][16 + half * 8];
    }
#pragma unroll
    for (int ni = 0; ni < 4; ++ni) {
      const int nc = wcol * 64 + ni * 16 + lr;
      // 16-bit B 32x16 layout: half 0 -> K{0..7,16..23}, half 1 -> K{8..15,24..31}
      bfr[ni].q[0] = *(const u32x4*)&BsT[nc][half * 8];
      bfr[ni].q[1] = *(const u32x4*)&BsT[nc][16 + half * 8];
    }
#pragma unroll
    for (int mi = 0; mi < 2; ++mi)
#pragma unroll
      for (int ni = 0; ni < 4; ++ni)
        acc[mi][ni] = wmma_bf16(afr[mi], bfr[ni], acc[mi][ni]);
    __syncthreads();
  }

  // Epilogue. C layout: VGPR r -> row r+8*half, col = lane&15.
#pragma unroll
  for (int mi = 0; mi < 2; ++mi) {
#pragma unroll
    for (int ni = 0; ni < 4; ++ni) {
#pragma unroll
      for (int r = 0; r < 8; ++r) {
        const long long row = bM + wrow * 32 + mi * 16 + r + 8 * half;
        const long long col = bN + wcol * 64 + ni * 16 + lr;
        const long long idx = row * (long long)N + col;
        const float vv = acc[mi][ni][r];
        if (EPI == EPI_BF16) {
          ((unsigned short*)Cout)[idx] = f2bf(vv);
        } else if (EPI == EPI_F32) {
          ((float*)Cout)[idx] = vv;
        } else if (EPI == EPI_GELU_MUL) {  // gelu(g, exact) * u
          const float uu = bf2f(((const unsigned short*)aux)[idx]);
          const float g  = 0.5f * vv * (1.0f + erff(vv * 0.70710678118654752f));
          ((unsigned short*)Cout)[idx] = f2bf(g * uu);
        } else {                            // residual add
          ((float*)Cout)[idx] = vv + ((const float*)aux)[idx];
        }
      }
    }
  }
}

// ---------------------------------------------------------------------------
// Flash-style attention (no causal mask in reference), softcap tanh.
// Block = 128 threads (4 waves); each wave owns a 16-query tile; block loops
// over 32-key tiles. K staged row-major via async DMA (QK^T B-frags are
// contiguous); V staged TRANSPOSED (VsT[col][key]) so P*V B-frags are also
// contiguous ds_load_b128. P converts C-layout -> A-layout via per-wave LDS.
// ---------------------------------------------------------------------------
__global__ __launch_bounds__(128) void attn_kernel(const unsigned short* __restrict__ Q,
                                                   const unsigned short* __restrict__ K,
                                                   const unsigned short* __restrict__ V,
                                                   unsigned short* __restrict__ O) {
  __shared__ unsigned short Ks[32][256];     // 16KB, row-major [key][dim]
  __shared__ unsigned short VsT[256][32];    // 16KB, transposed [dim][key]
  __shared__ unsigned short Ps[4][16][32];   // 4KB per-wave P scratch

  const int t = threadIdx.x;
  const int lane = t & 31, w = t >> 5;
  const int half = lane >> 4, lr = lane & 15;

  const int bh   = blockIdx.x >> 5;   // 32 q-blocks of 64 per (b,h)
  const int qblk = blockIdx.x & 31;
  const int b    = bh / kNH;
  const int hh   = bh % kNH;
  const int kvh  = hh >> 1;           // n_rep = NH/NKV = 2

  const long long qrow0 = (long long)b * kS + qblk * 64 + w * 16;

  // Q fragments for the 8 K-dim chunks of 32 (held for whole kernel)
  Frag qa[8];
  {
    const unsigned short* qp = Q + (qrow0 + lr) * (long long)(kNH * kHD) + hh * kHD;
#pragma unroll
    for (int kc = 0; kc < 8; ++kc) {
      qa[kc].q[0] = *(const u32x4*)(qp + kc * 32 + half * 8);
      qa[kc].q[1] = *(const u32x4*)(qp + kc * 32 + 16 + half * 8);
    }
  }

  f32x8 o[16];
#pragma unroll
  for (int nb = 0; nb < 16; ++nb)
#pragma unroll
    for (int r = 0; r < 8; ++r) o[nb][r] = 0.f;

  float rmax[8], rsum[8];
#pragma unroll
  for (int r = 0; r < 8; ++r) { rmax[r] = -3.0e38f; rsum[r] = 0.f; }

  const int srow = t >> 2;           // K staging: key 0..31
  const int scol = (t & 3) * 64;     //            64 dims per thread
  const long long kvbase = (long long)b * kS;
  const long long kvstr  = (long long)(kNKV * kHD);

  for (int kt = 0; kt < kS / 32; ++kt) {
    // --- K tile: async DMA, row-major (8 x 16B per thread)
    const unsigned short* kp = K + (kvbase + kt * 32 + srow) * kvstr + kvh * kHD + scol;
#pragma unroll
    for (int i = 0; i < 64; i += 8)
      async_copy_b128(&Ks[srow][scol + i], kp + i);
    // --- V tile: load row-major, store transposed (b16/d16_hi stores)
    const unsigned short* vp = V + (kvbase + kt * 32 + srow) * kvstr + kvh * kHD + scol;
#pragma unroll
    for (int i = 0; i < 64; i += 8) {
      const u32x4 d = *(const u32x4*)(vp + i);
#pragma unroll
      for (int e = 0; e < 4; ++e) {
        const unsigned wd = d[e];
        const int c = scol + i + 2 * e;
        VsT[c][srow]     = (unsigned short)wd;
        VsT[c + 1][srow] = (unsigned short)(wd >> 16);
      }
    }
    wait_async();
    __syncthreads();

    // scores for 32 keys: two 16x16 C tiles
    f32x8 s0, s1;
#pragma unroll
    for (int r = 0; r < 8; ++r) { s0[r] = 0.f; s1[r] = 0.f; }
#pragma unroll
    for (int kc = 0; kc < 8; ++kc) {
      Frag kb0, kb1;   // B frag: element(kk,n) = K[key n][kdim]
      kb0.q[0] = *(const u32x4*)&Ks[lr][kc * 32 + half * 8];
      kb0.q[1] = *(const u32x4*)&Ks[lr][kc * 32 + 16 + half * 8];
      kb1.q[0] = *(const u32x4*)&Ks[16 + lr][kc * 32 + half * 8];
      kb1.q[1] = *(const u32x4*)&Ks[16 + lr][kc * 32 + 16 + half * 8];
      s0 = wmma_bf16(qa[kc], kb0, s0);
      s1 = wmma_bf16(qa[kc], kb1, s1);
    }

    // softcap + online softmax (row = r+8*half, spread over 16-lane group)
    float alpha[8];
#pragma unroll
    for (int r = 0; r < 8; ++r) {
      float x0 = kSoftcap * fast_tanh(s0[r] * kCapMul);
      float x1 = kSoftcap * fast_tanh(s1[r] * kCapMul);
      float bm = fmaxf(x0, x1);
#pragma unroll
      for (int m = 1; m < 16; m <<= 1) bm = fmaxf(bm, __shfl_xor(bm, m, 32));
      const float mn = fmaxf(rmax[r], bm);
      alpha[r] = __expf(rmax[r] - mn);
      rmax[r]  = mn;
      const float p0 = __expf(x0 - mn);
      const float p1 = __expf(x1 - mn);
      s0[r] = p0; s1[r] = p1;
      float ps = p0 + p1;
#pragma unroll
      for (int m = 1; m < 16; m <<= 1) ps += __shfl_xor(ps, m, 32);
      rsum[r] = rsum[r] * alpha[r] + ps;
    }
#pragma unroll
    for (int nb = 0; nb < 16; ++nb)
#pragma unroll
      for (int r = 0; r < 8; ++r) o[nb][r] *= alpha[r];

    // P: C-layout -> LDS -> A-layout fragment (16x32)
#pragma unroll
    for (int r = 0; r < 8; ++r) {
      Ps[w][r + 8 * half][lr]      = f2bf(s0[r]);
      Ps[w][r + 8 * half][16 + lr] = f2bf(s1[r]);
    }
    Frag pf;   // per-wave LDS ops are in-order, no barrier needed
    pf.q[0] = *(const u32x4*)&Ps[w][lr][half * 8];
    pf.q[1] = *(const u32x4*)&Ps[w][lr][16 + half * 8];

    // O += P * V  (V B-frags contiguous from transposed tile)
#pragma unroll
    for (int nb = 0; nb < 16; ++nb) {
      Frag vf;   // element(kk,n) = V[key kk][col] = VsT[col][kk]
      const int c = nb * 16 + lr;
      vf.q[0] = *(const u32x4*)&VsT[c][half * 8];
      vf.q[1] = *(const u32x4*)&VsT[c][16 + half * 8];
      o[nb] = wmma_bf16(pf, vf, o[nb]);
    }
    __syncthreads();
  }

  float inv[8];
#pragma unroll
  for (int r = 0; r < 8; ++r) inv[r] = 1.0f / rsum[r];
#pragma unroll
  for (int nb = 0; nb < 16; ++nb) {
#pragma unroll
    for (int r = 0; r < 8; ++r) {
      const long long row = qrow0 + r + 8 * half;
      const int       col = hh * kHD + nb * 16 + lr;
      O[row * (long long)(kNH * kHD) + col] = f2bf(o[nb][r] * inv[r]);
    }
  }
}

// ---------------------------------------------------------------------------
// Launcher. Workspace layout (phase-disjoint aliasing, ~176MB total):
//   [0,32M)    h     f32 [4096,2048]
//   [32,48M)   xb    bf16 [4096,2048]   (attn-norm x; reused for MLP input m)
//   [48,112M)  r1:   y f32 [4096,2048] then u bf16 [4096,8192]
//   [112,176M) r2:   attn bf16 | q bf16 | k bf16 | v bf16, then act bf16 [4096,8192]
// ---------------------------------------------------------------------------
extern "C" void kernel_launch(void* const* d_in, const int* in_sizes, int n_in,
                              void* d_out, int out_size, void* d_ws, size_t ws_size,
                              hipStream_t stream) {
  (void)in_sizes; (void)n_in; (void)out_size; (void)ws_size;
  const int*   ids        = (const int*)d_in[0];
  const int*   pos        = (const int*)d_in[1];
  const float* emb        = (const float*)d_in[2];
  const float* Wq         = (const float*)d_in[3];
  const float* Wk         = (const float*)d_in[4];
  const float* Wv         = (const float*)d_in[5];
  const float* Wo         = (const float*)d_in[6];
  const float* Wg         = (const float*)d_in[7];
  const float* Wu         = (const float*)d_in[8];
  const float* Wd         = (const float*)d_in[9];
  const float* attn_norm  = (const float*)d_in[10];
  const float* mlp_norm   = (const float*)d_in[11];
  const float* mlp_norm2  = (const float*)d_in[12];
  const float* final_norm = (const float*)d_in[13];

  char* ws = (char*)d_ws;
  float*          h     = (float*)ws;
  unsigned short* xb    = (unsigned short*)(ws + 33554432LL);
  char*           r1    = ws + 33554432LL + 16777216LL;
  float*          y     = (float*)r1;                // dead before u written
  unsigned short* u     = (unsigned short*)r1;
  char*           r2    = r1 + 67108864LL;
  unsigned short* attnb = (unsigned short*)r2;       // dead before act written
  unsigned short* q     = (unsigned short*)(r2 + 16777216LL);
  unsigned short* k     = (unsigned short*)(r2 + 33554432LL);
  unsigned short* v     = (unsigned short*)(r2 + 41943040LL);
  unsigned short* act   = (unsigned short*)r2;

  embed_kernel<<<kTok, 256, 0, stream>>>(ids, emb, h);

  for (int l = 0; l < kL; ++l) {
    const float* wq = Wq + (long long)l * kHID * (kNH * kHD);
    const float* wk = Wk + (long long)l * kHID * (kNKV * kHD);
    const float* wv = Wv + (long long)l * kHID * (kNKV * kHD);
    const float* wo = Wo + (long long)l * (kNH * kHD) * kHID;
    const float* wg = Wg + (long long)l * kHID * kFF;
    const float* wu = Wu + (long long)l * kHID * kFF;
    const float* wd = Wd + (long long)l * kFF * kHID;

    rmsnorm_kernel<0><<<kTok, 256, 0, stream>>>(h, attn_norm + l * kHID, xb);

    gemm_kernel<EPI_BF16><<<dim3(16, 32), 256, 0, stream>>>(xb, wq, q, nullptr, kTok, 2048, 2048);
    gemm_kernel<EPI_BF16><<<dim3( 8, 32), 256, 0, stream>>>(xb, wk, k, nullptr, kTok, 1024, 2048);
    gemm_kernel<EPI_BF16><<<dim3( 8, 32), 256, 0, stream>>>(xb, wv, v, nullptr, kTok, 1024, 2048);

    rope_kernel<<<kTok, 256, 0, stream>>>(q, k, pos);

    attn_kernel<<<kB * kNH * (kS / 64), 128, 0, stream>>>(q, k, v, attnb);

    gemm_kernel<EPI_F32><<<dim3(16, 32), 256, 0, stream>>>(attnb, wo, y, nullptr, kTok, 2048, 2048);

    rmsnorm_kernel<2><<<kTok, 256, 0, stream>>>(y, mlp_norm + l * kHID, h);    // h = attn + rms(attn)
    rmsnorm_kernel<0><<<kTok, 256, 0, stream>>>(h, mlp_norm2 + l * kHID, xb);  // m

    gemm_kernel<EPI_BF16>    <<<dim3(64, 32), 256, 0, stream>>>(xb, wu, u,   nullptr, kTok, kFF, 2048);
    gemm_kernel<EPI_GELU_MUL><<<dim3(64, 32), 256, 0, stream>>>(xb, wg, act, u,       kTok, kFF, 2048);
    gemm_kernel<EPI_ADD_F32> <<<dim3(16, 32), 256, 0, stream>>>(act, wd, h,  h,       kTok, 2048, kFF);
  }

  rmsnorm_kernel<1><<<kTok, 256, 0, stream>>>(h, final_norm, (float*)d_out);
}
